// GraphAttentionLayer_56444460204319
// MI455X (gfx1250) — matrix operand order repaired and checked by reading, not compiled, not run on hardware
//
#include <hip/hip_runtime.h>
#include <hip/hip_bf16.h>

typedef __attribute__((ext_vector_type(16))) _Float16 v16h;
typedef __attribute__((ext_vector_type(8)))  _Float16 v8h;
typedef __attribute__((ext_vector_type(8)))  float    v8f;

#define N_NODES 8192
#define FIN     1024
#define FOUT    256
#define ALPHA   0.2f
#define LPAD    40   // kernel-1 LDS col stride in halves (80B = 20 dwords)
#define HPAD    72   // kernel-3 LDS col stride in halves (144B = 36 dwords)

union AFrag { v16h v; v8h h[2]; };

#if defined(__has_builtin)
#  if __has_builtin(__builtin_amdgcn_ballot_w32)
#    define HAVE_BALLOT32 1
#  endif
#  if __has_builtin(__builtin_amdgcn_s_wait_asynccnt)
#    define HAVE_WAIT_ASYNC 1
#  endif
#endif

__device__ __forceinline__ unsigned wave_ballot32(bool pred, int lane) {
#ifdef HAVE_BALLOT32
    return __builtin_amdgcn_ballot_w32(pred);
#else
    unsigned bit = pred ? (1u << lane) : 0u;
    #pragma unroll
    for (int off = 16; off >= 1; off >>= 1) bit |= __shfl_xor(bit, off, 32);
    return bit;
#endif
}

__device__ __forceinline__ void wait_asynccnt0() {
#ifdef HAVE_WAIT_ASYNC
    __builtin_amdgcn_s_wait_asynccnt(0);
#else
    asm volatile("s_wait_asynccnt 0x0" ::: "memory");
#endif
}

// async copy 16B: global (per-lane 64-bit addr) -> LDS (per-lane byte addr)
__device__ __forceinline__ void async_g2l_b128(unsigned lds_addr, uint64_t gaddr) {
    asm volatile("global_load_async_to_lds_b128 %0, %1, off"
                 :: "v"(lds_addr), "v"(gaddr) : "memory");
}

// =====================================================================
// Kernel 1: hT = f16( X[8192,1024] @ W[1024,256] ) stored TRANSPOSED
// as hT[col][row] (256 x 8192). Epilogue: one contiguous v8h store per
// tile per lane.
// =====================================================================
__global__ void __launch_bounds__(256)
gat_gemm_h(const float* __restrict__ X, const float* __restrict__ W,
           _Float16* __restrict__ hT) {
    __shared__ __align__(16) _Float16 ldsX[64 * LPAD];       // [row][k], 5 KB
    __shared__ __align__(16) unsigned ldsWTu[64 * (LPAD/2)]; // [col][k], 5 KB
    _Float16* ldsWT = (_Float16*)ldsWTu;

    const int tid  = threadIdx.x;
    const int lane = tid & 31;
    const int wave = tid >> 5;
    const int myn  = lane & 15;
    const int khalf8  = ((lane >> 4) & 1) * 8;
    const int khalf16 = ((lane >> 4) & 1) * 16;
    const int row0 = blockIdx.x * 64;
    const int col0 = blockIdx.y * 64;
    const int tr   = wave & 3;          // row tile 0..3
    const int tc0  = (wave >> 2) * 2;   // first of 2 col tiles

    v8f acc0 = {}; v8f acc1 = {};

    for (int k0 = 0; k0 < FIN; k0 += 32) {
        __syncthreads();
        #pragma unroll
        for (int ii = 0; ii < 8; ++ii) {
            int idx = tid + ii * 256;                 // 0..2047
            int r = idx >> 5, c = idx & 31;
            ldsX[r * LPAD + c] = (_Float16)X[(size_t)(row0 + r) * FIN + k0 + c];
        }
        #pragma unroll
        for (int ii = 0; ii < 4; ++ii) {
            int idx = tid + ii * 256;                 // 0..1023
            int c  = idx & 63;
            int rp = idx >> 6;                        // row-pair 0..15
            _Float16 h0 = (_Float16)W[(size_t)(k0 + 2 * rp) * FOUT + col0 + c];
            _Float16 h1 = (_Float16)W[(size_t)(k0 + 2 * rp + 1) * FOUT + col0 + c];
            union { unsigned u; _Float16 f[2]; } pk;
            pk.f[0] = h0; pk.f[1] = h1;
            ldsWTu[c * (LPAD/2) + rp] = pk.u;
        }
        __syncthreads();

        AFrag A;
        A.h[0] = *(const v8h*)&ldsX[(tr * 16 + myn) * LPAD + khalf16];
        A.h[1] = *(const v8h*)&ldsX[(tr * 16 + myn) * LPAD + khalf16 + 8];
        v16h B0 = *(const v16h*)&ldsWT[(tc0 * 16 + myn) * LPAD + khalf16];
        v16h B1 = *(const v16h*)&ldsWT[((tc0 + 1) * 16 + myn) * LPAD + khalf16];

        acc0 = __builtin_amdgcn_wmma_f32_16x16x32_f16(false, A.v, false, B0,
                                                      (short)0, acc0, false, false);
        acc1 = __builtin_amdgcn_wmma_f32_16x16x32_f16(false, A.v, false, B1,
                                                      (short)0, acc1, false, false);
    }

    // epilogue: lane owns column (tc*16+myn), rows (tr*16 + m0 .. +7) contiguous
    const int m0 = (lane >> 4) ? 8 : 0;
    v8h o0, o1;
    #pragma unroll
    for (int r = 0; r < 8; ++r) { o0[r] = (_Float16)acc0[r]; o1[r] = (_Float16)acc1[r]; }
    const size_t rowb = (size_t)row0 + tr * 16 + m0;
    *(v8h*)&hT[(size_t)(col0 + tc0 * 16 + myn) * N_NODES + rowb]       = o0;
    *(v8h*)&hT[(size_t)(col0 + (tc0 + 1) * 16 + myn) * N_NODES + rowb] = o1;
}

// =====================================================================
// Kernel 2: f_src[i] = h[i,:]·a1 ; f_dst[i] = h[i,:]·a2
// thread-per-row over transposed hT -> fully coalesced, no reductions.
// =====================================================================
__global__ void __launch_bounds__(256)
gat_fvec(const _Float16* __restrict__ hT, const float* __restrict__ a,
         float* __restrict__ fsrc, float* __restrict__ fdst) {
    const int row = blockIdx.x * 256 + threadIdx.x;
    float s1 = 0.f, s2 = 0.f;
    #pragma unroll 8
    for (int c = 0; c < FOUT; ++c) {
        float hv = (float)hT[(size_t)c * N_NODES + row];
        s1 += hv * a[c];
        s2 += hv * a[FOUT + c];
    }
    fsrc[row] = s1;
    fdst[row] = s2;
}

// =====================================================================
// Kernel 3: per 16-row block: masked softmax stats (adj read ONCE from
// HBM, bits in LDS), then out = elu( softmax(att) @ h ) via WMMA.
// h tile staged with ASYNC global->LDS b128 copies (per-lane column),
// overlapped with p-tile exp computation; K=64 per round, 4 WMMAs/wave.
// =====================================================================
__global__ void __launch_bounds__(256)
gat_attn(const int* __restrict__ adj, const _Float16* __restrict__ hT,
         const float* __restrict__ fsrc, const float* __restrict__ fdst,
         float* __restrict__ out) {
    __shared__ unsigned ldsBits[16 * 256];                    // 16 KB adjacency bits
    __shared__ __align__(16) _Float16 ldsHT[256 * HPAD];      // h tile [col][k0..63], 36 KB
    __shared__ __align__(16) _Float16 ldsP[16 * HPAD];        // p tile [row][k0..63], 2.25 KB
    __shared__ float ldsM[16], ldsInvS[16], ldsFs[16];

    const int tid  = threadIdx.x;
    const int lane = tid & 31;
    const int wave = tid >> 5;
    const int myn  = lane & 15;
    const int khalf8  = ((lane >> 4) & 1) * 8;
    const int khalf16 = ((lane >> 4) & 1) * 16;
    const int row0 = blockIdx.x * 16;

    if (tid < 16) ldsFs[tid] = fsrc[row0 + tid];

    // ---------- Phase A: online softmax stats + adj bit-pack ----------
    #pragma unroll
    for (int rr = 0; rr < 2; ++rr) {
        const int lr  = wave * 2 + rr;
        const int row = row0 + lr;
        const float fs = fsrc[row];
        const int* arow = adj + (size_t)row * N_NODES;
        float m = -3.0e38f, s = 0.f;
        for (int t = 0; t < N_NODES / 32; ++t) {
            int j  = t * 32 + lane;
            __builtin_prefetch(arow + j + 256, 0, 1);   // global_prefetch_b8
            int av = arow[j];
            unsigned word = wave_ballot32(av > 0, lane);
            if (lane == 0) ldsBits[lr * 256 + t] = word;
            if (av > 0) {
                float e = fs + fdst[j];
                e = fmaxf(e, ALPHA * e);                // leaky relu (alpha<1)
                float M = fmaxf(m, e);
                s = s * __expf(m - M) + __expf(e - M);
                m = M;
            }
        }
        #pragma unroll
        for (int off = 16; off >= 1; off >>= 1) {
            float m2 = __shfl_xor(m, off, 32);
            float s2 = __shfl_xor(s, off, 32);
            float M  = fmaxf(m, m2);
            s = s * __expf(m - M) + s2 * __expf(m2 - M);
            m = M;
        }
        if (lane == 0) {
            ldsM[lr]    = m;
            ldsInvS[lr] = (s > 0.f) ? (1.0f / s) : 0.f;
        }
    }
    __syncthreads();

    // ---------- Phase B: K=64 per round, async-staged h tile ----------
    v8f acc0 = {}, acc1 = {};
    // this thread owns h column `tid`: 128 contiguous bytes per round
    const unsigned myLdsCol = (unsigned)(uintptr_t)&ldsHT[tid * HPAD];
    const _Float16* myGCol  = hT + (size_t)tid * N_NODES;

    for (int j0 = 0; j0 < N_NODES; j0 += 64) {
        __syncthreads();                       // previous tile fully consumed
        // issue async copies: col tid, k = j0..j0+63 (8 x 16B)
        {
            uint64_t ga = (uint64_t)(uintptr_t)(myGCol + j0);
            #pragma unroll
            for (int ii = 0; ii < 8; ++ii)
                async_g2l_b128(myLdsCol + ii * 16, ga + ii * 16);
        }
        // build p tile: 16x64 f16, 4 per thread (overlaps async copies)
        {
            int idx = tid * 4;
            int i   = idx >> 6;
            int jj  = idx & 63;
            unsigned word = ldsBits[i * 256 + (j0 >> 5) + (jj >> 5)];
            float fs = ldsFs[i];
            float mi = ldsM[i];
            #pragma unroll
            for (int e2 = 0; e2 < 4; ++e2) {
                int j = jj + e2;
                float p = 0.f;
                if ((word >> (j & 31)) & 1u) {
                    float ev = fs + fdst[j0 + j];
                    ev = fmaxf(ev, ALPHA * ev);
                    p  = __expf(ev - mi);
                }
                ldsP[i * HPAD + j] = (_Float16)p;
            }
        }
        wait_asynccnt0();                      // h tile landed in LDS
        __syncthreads();                       // visible to all waves

        AFrag A0, A1;
        A0.h[0] = *(const v8h*)&ldsP[myn * HPAD + khalf16];
        A0.h[1] = *(const v8h*)&ldsP[myn * HPAD + khalf16 + 8];
        A1.h[0] = *(const v8h*)&ldsP[myn * HPAD + 32 + khalf16];
        A1.h[1] = *(const v8h*)&ldsP[myn * HPAD + 40 + khalf16];

        const _Float16* b0p = &ldsHT[(wave * 32 + myn) * HPAD];
        const _Float16* b1p = &ldsHT[(wave * 32 + 16 + myn) * HPAD];
        v16h B00 = *(const v16h*)&b0p[khalf16];
        v16h B10 = *(const v16h*)&b1p[khalf16];
        v16h B01 = *(const v16h*)&b0p[32 + khalf16];
        v16h B11 = *(const v16h*)&b1p[32 + khalf16];

        acc0 = __builtin_amdgcn_wmma_f32_16x16x32_f16(false, A0.v, false, B00,
                                                      (short)0, acc0, false, false);
        acc1 = __builtin_amdgcn_wmma_f32_16x16x32_f16(false, A0.v, false, B10,
                                                      (short)0, acc1, false, false);
        acc0 = __builtin_amdgcn_wmma_f32_16x16x32_f16(false, A1.v, false, B01,
                                                      (short)0, acc0, false, false);
        acc1 = __builtin_amdgcn_wmma_f32_16x16x32_f16(false, A1.v, false, B11,
                                                      (short)0, acc1, false, false);
    }

    // ---------- Epilogue: scale by 1/sum, ELU, store f32 ----------
    const int m0 = (lane >> 4) ? 8 : 0;
    #pragma unroll
    for (int r = 0; r < 8; ++r) {
        int lr = m0 + r;
        float inv = ldsInvS[lr];
        float x0 = acc0[r] * inv;
        float x1 = acc1[r] * inv;
        x0 = (x0 > 0.f) ? x0 : (__expf(x0) - 1.f);
        x1 = (x1 > 0.f) ? x1 : (__expf(x1) - 1.f);
        size_t o = (size_t)(row0 + lr) * FOUT + wave * 32 + myn;
        out[o]      = x0;
        out[o + 16] = x1;
    }
}

// =====================================================================
extern "C" void kernel_launch(void* const* d_in, const int* in_sizes, int n_in,
                              void* d_out, int out_size, void* d_ws, size_t ws_size,
                              hipStream_t stream) {
    const float* X   = (const float*)d_in[0];   // [8192,1024]
    const int*   adj = (const int*)d_in[1];     // [8192,8192]
    const float* W   = (const float*)d_in[2];   // [1024,256]
    const float* a   = (const float*)d_in[3];   // [512,1]
    float*       out = (float*)d_out;           // [8192,256]

    char* ws = (char*)d_ws;
    _Float16* hT   = (_Float16*)ws;                              // 4 MB, [256][8192]
    float*    fsrc = (float*)(ws + (size_t)N_NODES * FOUT * 2);  // 32 KB
    float*    fdst = fsrc + N_NODES;                             // 32 KB

    dim3 blk(256);
    gat_gemm_h<<<dim3(N_NODES / 64, FOUT / 64), blk, 0, stream>>>(X, W, hT);
    gat_fvec  <<<dim3(N_NODES / 256),           blk, 0, stream>>>(hT, a, fsrc, fdst);
    gat_attn  <<<dim3(N_NODES / 16),            blk, 0, stream>>>(adj, hT, fsrc, fdst, out);
}
